// Attention_644245094761
// MI455X (gfx1250) — matrix operand order repaired and checked by reading, not compiled, not run on hardware
//
#include <hip/hip_runtime.h>
#include <hip/hip_bf16.h>

// MHA: B=8, N=1024, D=768, H=12, HD=64.  fp32 I/O, f16 WMMA compute, fp32 accum.
#define BB   8
#define NN   1024
#define DD   768
#define HH   12
#define HDIM 64
#define SCALE 0.125f            // HD^-0.5

typedef __attribute__((ext_vector_type(16))) _Float16 v16h;
typedef __attribute__((ext_vector_type(8)))  float    v8f;
typedef __attribute__((ext_vector_type(4)))  _Float16 v4h;

union Frag { v16h v; _Float16 h[16]; uint4 q[2]; };

// A-matrix (16x32 f16) per ISA 7.12.2: lane = row M; per lane two contiguous
// 8-element chunks:  k in [hi, hi+8) and [16+hi, 16+hi+8), hi = (lane&16)?8:0.
__device__ __forceinline__ void load_a(Frag& a, const _Float16* rowk0, int lane) {
  int hi = (lane & 16) ? 8 : 0;
  a.q[0] = *(const uint4*)(rowk0 + hi);
  a.q[1] = *(const uint4*)(rowk0 + 16 + hi);
}
// B-matrix (32x16 f16): lane = column N; same k order as A: two 8-element
// chunks k in [hi, hi+8) and [16+hi, 16+hi+8), hi = (lane&16)?8:0.
__device__ __forceinline__ void load_b(Frag& b, const _Float16* colk0, int lane) {
  int hi = (lane & 16) ? 8 : 0;
  b.q[0] = *(const uint4*)(colk0 + hi);
  b.q[1] = *(const uint4*)(colk0 + 16 + hi);
}

__device__ __forceinline__ v8f wmma_f16(const Frag& a, const Frag& b, v8f c) {
  return __builtin_amdgcn_wmma_f32_16x16x32_f16(false, a.v, false, b.v,
                                                (short)0, c, false, false);
}

// ---------------------------------------------------------------------------
// Prep 1: x (fp32) -> xh (f16), elementwise, vectorized x4.
// ---------------------------------------------------------------------------
__global__ __launch_bounds__(256) void cvt_x_kernel(
    const float* __restrict__ x, _Float16* __restrict__ xh)
{
  int i = blockIdx.x * 256 + threadIdx.x;
  float4 v = ((const float4*)x)[i];
  v4h o;
  o.x = (_Float16)v.x; o.y = (_Float16)v.y;
  o.z = (_Float16)v.z; o.w = (_Float16)v.w;
  ((v4h*)xh)[i] = o;
}

// ---------------------------------------------------------------------------
// Prep 2: w [K][Ncol] fp32  ->  wT [Ncol][K] f16   (32x32 LDS tile transpose)
// grid = (Ncol/32, K/32), block = 256 (32x8)
// ---------------------------------------------------------------------------
__global__ __launch_bounds__(256) void cvt_transpose_kernel(
    const float* __restrict__ w, _Float16* __restrict__ wT, int K, int Ncol)
{
  __shared__ float tile[32][33];
  int tx = threadIdx.x & 31, ty = threadIdx.x >> 5;
  int n0 = blockIdx.x * 32, k0 = blockIdx.y * 32;
#pragma unroll
  for (int i = 0; i < 4; ++i)
    tile[ty + i * 8][tx] = w[(size_t)(k0 + ty + i * 8) * Ncol + n0 + tx];
  __syncthreads();
#pragma unroll
  for (int i = 0; i < 4; ++i)
    wT[(size_t)(n0 + ty + i * 8) * K + k0 + tx] = (_Float16)tile[tx][ty + i * 8];
}

// ---------------------------------------------------------------------------
// 32x64 GEMM core: acc[g][t] += A(32xDD) * B(DDx64); two A row-groups share
// the four B fragments -> 8 WMMAs per 12 b128 loads per k-step.
// ---------------------------------------------------------------------------
__device__ __forceinline__ void gemm_32x64(
    v8f acc[2][4], const _Float16* arow0, const _Float16* arow1,
    const _Float16* const bp[4], int lane)
{
  for (int k0 = 0; k0 < DD; k0 += 32) {
    Frag a0, a1;
    load_a(a0, arow0 + k0, lane);
    load_a(a1, arow1 + k0, lane);
#pragma unroll
    for (int t = 0; t < 4; ++t) {
      Frag b;
      load_b(b, bp[t] + k0, lane);
      acc[0][t] = wmma_f16(a0, b, acc[0][t]);
      acc[1][t] = wmma_f16(a1, b, acc[1][t]);
    }
  }
}

// ---------------------------------------------------------------------------
// Kernel 1: QKV = xh @ w_qkv + b_qkv -> Q (pre-scaled), K in [B,H,N,HD];
//           V stored transposed [B,H,HD,N].  Wave = 32(M) x 64(N) tile.
// ---------------------------------------------------------------------------
__global__ __launch_bounds__(256) void qkv_gemm_kernel(
    const _Float16* __restrict__ xh, const _Float16* __restrict__ wT,
    const float* __restrict__ bias,
    _Float16* __restrict__ qb, _Float16* __restrict__ kbuf,
    _Float16* __restrict__ vb)
{
  const int lane = threadIdx.x & 31, wave = threadIdx.x >> 5;
  const int NT = (3 * DD) / 64;                 // 36 column tiles
  int wt = blockIdx.x * 8 + wave;               // 0..9215
  int mt = wt / NT, nt = wt % NT;               // mt: 0..255 (32-row tiles)
  int row0 = mt * 32, col0 = nt * 64;
  int m = lane & 15;

  const _Float16* arow0 = xh + (size_t)(row0 + m) * DD;
  const _Float16* arow1 = xh + (size_t)(row0 + 16 + m) * DD;
  const _Float16* bp[4];
#pragma unroll
  for (int t = 0; t < 4; ++t)
    bp[t] = wT + (size_t)(col0 + t * 16 + m) * DD;

  v8f acc[2][4] = {};
  gemm_32x64(acc, arow0, arow1, bp, lane);

  // col0 = which*768 + h*64  (64-wide tile == one head of one of q/k/v)
  int which = nt / HH, h = nt % HH;
  float scl = (which == 0) ? SCALE : 1.0f;      // fold softmax scale into Q
#pragma unroll
  for (int g = 0; g < 2; ++g)
#pragma unroll
    for (int t = 0; t < 4; ++t) {
#pragma unroll
      for (int r = 0; r < 8; ++r) {
        int mm  = (lane < 16) ? r : (8 + r);
        int row = row0 + g * 16 + mm;            // flat b*N + n
        int d   = t * 16 + m;
        float val = (acc[g][t][r] + bias[col0 + d]) * scl;
        int b_ = row >> 10, n_ = row & (NN - 1);
        size_t bh = (size_t)b_ * HH + h;
        if (which == 2)                          // V transposed: [B,H,HD,N]
          vb[(bh * HDIM + d) * NN + n_] = (_Float16)val;
        else if (which == 1)
          kbuf[(bh * NN + n_) * HDIM + d] = (_Float16)val;
        else
          qb[(bh * NN + n_) * HDIM + d] = (_Float16)val;
      }
    }
}

// ---------------------------------------------------------------------------
// Kernel 2: flash attention.  Block = (b,h, 128 q-rows); wave = 16 q-rows.
// All K/V fragment loads issued at iteration top so V-load latency hides
// behind score WMMAs + softmax.
// ---------------------------------------------------------------------------
__global__ __launch_bounds__(256) void attn_kernel(
    const _Float16* __restrict__ qb, const _Float16* __restrict__ kb,
    const _Float16* __restrict__ vt, _Float16* __restrict__ ctx)
{
  __shared__ __align__(16) _Float16 lds[8][16][32];   // per-wave P staging
  const int lane = threadIdx.x & 31, wave = threadIdx.x >> 5;
  int qblk = blockIdx.x & 7;                    // N/128 = 8
  int bh   = blockIdx.x >> 3;                   // b*H + h
  const _Float16* qp = qb + (size_t)bh * NN * HDIM;
  const _Float16* kp = kb + (size_t)bh * NN * HDIM;
  const _Float16* vp = vt + (size_t)bh * HDIM * NN;   // [HD][N]
  int q0 = qblk * 128 + wave * 16;
  int m  = lane & 15;

  Frag aq[2];                                   // Q rows (pre-scaled), 2 k-steps
  load_a(aq[0], qp + (size_t)(q0 + m) * HDIM, lane);
  load_a(aq[1], qp + (size_t)(q0 + m) * HDIM + 32, lane);

  // incremented base pointers: kill per-iteration address math
  const _Float16* kbase[2];
  const _Float16* vbase[4];
  kbase[0] = kp + (size_t)m * HDIM;             // key row m
  kbase[1] = kp + (size_t)(16 + m) * HDIM;      // key row 16+m
#pragma unroll
  for (int t = 0; t < 4; ++t)
    vbase[t] = vp + (size_t)(t * 16 + m) * NN;  // d row

  float mrow[8], lrow[8];
  v8f acc[4] = {};
#pragma unroll
  for (int r = 0; r < 8; ++r) { mrow[r] = -3.0e38f; lrow[r] = 0.f; }

  for (int it = 0; it < NN / 32; ++it) {
    // ---- issue ALL loads for this key block up front ----
    Frag kf[2][2], vf[4];
#pragma unroll
    for (int t = 0; t < 2; ++t) {
      load_b(kf[t][0], kbase[t], lane);         // d 0..31
      load_b(kf[t][1], kbase[t] + 32, lane);    // d 32..63
    }
#pragma unroll
    for (int t = 0; t < 4; ++t)
      load_b(vf[t], vbase[t], lane);
    kbase[0] += 32 * HDIM; kbase[1] += 32 * HDIM;
#pragma unroll
    for (int t = 0; t < 4; ++t) vbase[t] += 32;

    // ---- scores: two 16x16 tiles ----
    v8f s[2];
#pragma unroll
    for (int t = 0; t < 2; ++t) {
      v8f c = {};
      c = wmma_f16(aq[0], kf[t][0], c);
      c = wmma_f16(aq[1], kf[t][1], c);
      s[t] = c;
    }
    // ---- online softmax (lanes 0-15 own rows 0-7, lanes 16-31 rows 8-15) ----
    float p0[8], p1[8];
#pragma unroll
    for (int r = 0; r < 8; ++r) {
      float s0 = s[0][r];
      float s1 = s[1][r];
      float tm = fmaxf(s0, s1);
#pragma unroll
      for (int off = 8; off; off >>= 1)
        tm = fmaxf(tm, __shfl_xor(tm, off, 16));
      float nm    = fmaxf(mrow[r], tm);
      float alpha = __expf(mrow[r] - nm);
      float e0 = __expf(s0 - nm);
      float e1 = __expf(s1 - nm);
      float rs = e0 + e1;
#pragma unroll
      for (int off = 8; off; off >>= 1)
        rs += __shfl_xor(rs, off, 16);
      lrow[r] = lrow[r] * alpha + rs;
      mrow[r] = nm;
      p0[r] = e0; p1[r] = e1;
#pragma unroll
      for (int t = 0; t < 4; ++t) acc[t][r] *= alpha;
    }
    // ---- relayout P: C-layout -> A-layout via per-wave LDS (DS in-order) ----
#pragma unroll
    for (int r = 0; r < 8; ++r) {
      int mm = (lane < 16) ? r : (8 + r);
      lds[wave][mm][m]      = (_Float16)p0[r];
      lds[wave][mm][16 + m] = (_Float16)p1[r];
    }
    Frag pa;
    {
      const uint4* lp = (const uint4*)(&lds[wave][m][0]);
      int qi = (lane & 16) ? 1 : 0;
      pa.q[0] = lp[qi];
      pa.q[1] = lp[2 + qi];
    }
    // ---- O += P(16x32) x V(32x64) ----
#pragma unroll
    for (int t = 0; t < 4; ++t)
      acc[t] = wmma_f16(pa, vf[t], acc[t]);
  }
  // ---- normalize and write ctx [B, N, D] (heads re-interleaved) ----
  int b_ = bh / HH, h = bh % HH;
#pragma unroll
  for (int t = 0; t < 4; ++t)
#pragma unroll
    for (int r = 0; r < 8; ++r) {
      int mm = (lane < 16) ? r : (8 + r);
      float val = acc[t][r] / lrow[r];
      ctx[((size_t)(b_ * NN + q0 + mm)) * DD + h * HDIM + t * 16 + m] =
          (_Float16)val;
    }
}

// ---------------------------------------------------------------------------
// Kernel 3: out = ctx @ w_proj + b_proj   (fp32 output), 32x64 wave tiles.
// ---------------------------------------------------------------------------
__global__ __launch_bounds__(256) void proj_gemm_kernel(
    const _Float16* __restrict__ ctx, const _Float16* __restrict__ wT,
    const float* __restrict__ bias, float* __restrict__ out)
{
  const int lane = threadIdx.x & 31, wave = threadIdx.x >> 5;
  const int NT = DD / 64;                       // 12
  int wt = blockIdx.x * 8 + wave;               // 0..3071
  int mt = wt / NT, nt = wt % NT;               // mt: 0..255
  int row0 = mt * 32, col0 = nt * 64;
  int m = lane & 15;

  const _Float16* arow0 = ctx + (size_t)(row0 + m) * DD;
  const _Float16* arow1 = ctx + (size_t)(row0 + 16 + m) * DD;
  const _Float16* bp[4];
#pragma unroll
  for (int t = 0; t < 4; ++t)
    bp[t] = wT + (size_t)(col0 + t * 16 + m) * DD;

  v8f acc[2][4] = {};
  gemm_32x64(acc, arow0, arow1, bp, lane);

#pragma unroll
  for (int g = 0; g < 2; ++g)
#pragma unroll
    for (int t = 0; t < 4; ++t)
#pragma unroll
      for (int r = 0; r < 8; ++r) {
        int mm  = (lane < 16) ? r : (8 + r);
        int col = col0 + t * 16 + m;
        out[(size_t)(row0 + g * 16 + mm) * DD + col] = acc[g][t][r] + bias[col];
      }
}

// ---------------------------------------------------------------------------
extern "C" void kernel_launch(void* const* d_in, const int* in_sizes, int n_in,
                              void* d_out, int out_size, void* d_ws, size_t ws_size,
                              hipStream_t stream) {
  const float* x      = (const float*)d_in[0];
  const float* w_qkv  = (const float*)d_in[1];
  const float* b_qkv  = (const float*)d_in[2];
  const float* w_proj = (const float*)d_in[3];
  const float* b_proj = (const float*)d_in[4];
  float* out = (float*)d_out;

  const size_t SZ = (size_t)BB * HH * NN * HDIM;      // 6,291,456 f16 elems
  _Float16* ws     = (_Float16*)d_ws;
  _Float16* qb     = ws;
  _Float16* kb     = ws + SZ;
  _Float16* vt     = ws + 2 * SZ;                     // V transposed [B,H,HD,N]
  _Float16* ctx    = ws + 3 * SZ;
  _Float16* xh     = ws + 4 * SZ;                     // x as f16, same size
  _Float16* wqkvT  = ws + 5 * SZ;                     // [2304][768] f16
  _Float16* wprojT = wqkvT + (size_t)(3 * DD) * DD;   // [768][768]  f16

  // prep: fp32 -> f16 (+ weight transpose); ~16 MB of one-off traffic
  cvt_x_kernel<<<(BB * NN * DD) / 4 / 256, 256, 0, stream>>>(x, xh);
  cvt_transpose_kernel<<<dim3((3 * DD) / 32, DD / 32), 256, 0, stream>>>(
      w_qkv, wqkvT, DD, 3 * DD);
  cvt_transpose_kernel<<<dim3(DD / 32, DD / 32), 256, 0, stream>>>(
      w_proj, wprojT, DD, DD);

  // 8192/32 * 2304/64 wave-tiles / 8 waves = 1152 blocks
  qkv_gemm_kernel<<<1152, 256, 0, stream>>>(xh, wqkvT, b_qkv, qb, kb, vt);
  // B*H*(N/128) = 768 blocks
  attn_kernel<<<768, 256, 0, stream>>>(qb, kb, vt, ctx);
  // 8192/32 * 768/64 / 8 = 384 blocks
  proj_gemm_kernel<<<384, 256, 0, stream>>>(ctx, wprojT, b_proj, out);
}
